// GSIGNN_50568944943197
// MI455X (gfx1250) — matrix operand order repaired and checked by reading, not compiled, not run on hardware
//
#include <hip/hip_runtime.h>

#define N_NODE 2048
#define DIM    256
#define NH     8
#define DK     32
#define LN_EPS 1e-5f

typedef __attribute__((ext_vector_type(16))) _Float16 v16h;
typedef __attribute__((ext_vector_type(8)))  _Float16 v8h;
typedef __attribute__((ext_vector_type(8)))  float    v8f;
typedef __attribute__((ext_vector_type(4)))  unsigned int v4u;
typedef __attribute__((ext_vector_type(8)))  int      v8i;
typedef __attribute__((ext_vector_type(4)))  int      v4i;

#if defined(__has_builtin)
# if __has_builtin(__builtin_amdgcn_tensor_load_to_lds)
#  define HAVE_TDM 1
# endif
#endif

union AFrag { v16h v; v8h h[2]; };

// A-matrix (16x32 f16) fragment, contiguous k order (B uses the same):
// lane L: row m = L&15 ; kb = (L>>4)*16 ; elems 0..15 = K kb..kb+15
__device__ inline v16h load_afrag(const _Float16* base_kb) {
  AFrag f;
  f.h[0] = *(const v8h*)(base_kb);
  f.h[1] = *(const v8h*)(base_kb + 8);
  return f.v;
}

// ---------------------------------------------------------------------------
// Generic GEMM: C[n,m] = act( sum_k X[n,k]*W[m,k] + bias[m] )
// X: N_NODE x K (f16, ld=ldx), W: M x K (f16, ld=ldw, with ldw==K for all uses).
// Block = 4 waves; wave computes a 16x64 tile. The block's 64-row B panel is
// DMA'd once into LDS by the Tensor Data Mover, then all B fragments are LDS
// broadcasts; the streaming A operand stays on direct global_load_b128.
// act: 0=none 1=relu 2=sigmoid 3=tanh
// ---------------------------------------------------------------------------
__global__ __launch_bounds__(128)
void gemm_xwt(const _Float16* __restrict__ X, int ldx,
              const _Float16* __restrict__ W, int ldw, int M,
              const float* __restrict__ bias,
              float* __restrict__ Cf, int ldc,
              _Float16* __restrict__ Ch, int ldch,
              int K, int act)
{
  extern __shared__ __attribute__((aligned(64))) _Float16 bpanel[];   // 64 x K

  const int lane = threadIdx.x & 31;
  const int wave = threadIdx.x >> 5;
  const int row0 = blockIdx.y * 64 + wave * 16;
  const int col0 = blockIdx.x * 64;
  const int m    = lane & 15;
  const int hi   = lane >> 4;
  const int kbA  = hi * 16;
  const int kbB  = hi * 16;

#ifdef HAVE_TDM
  if (threadIdx.x < 32) {
    const _Float16* gsrc = W + (size_t)col0 * ldw;        // tile start (64 rows x K)
    unsigned long long ga = (unsigned long long)(size_t)gsrc;
    unsigned lds_off = (unsigned)(size_t)(&bpanel[0]);
    // D# group0: count=1 | lds_addr | global_addr[56:0] | type=2
    v4u g0 = { 1u, lds_off, (unsigned)ga,
               (unsigned)((ga >> 32) & 0x01ffffffu) | 0x80000000u };
    // D# group1: data_size=2B, tensor 2D (K x M, stride ldw), tile K x 64
    v8i g1;
    g1[0] = 0x00010000;                                               // data_size=1 -> 2 bytes
    g1[1] = (int)(((unsigned)K & 0xffffu) << 16);                     // tensor_dim0 lo16
    g1[2] = (int)(((unsigned)K >> 16) | (((unsigned)M & 0xffffu) << 16)); // dim0 hi / dim1 lo
    g1[3] = (int)(((unsigned)M >> 16) | (((unsigned)K & 0xffffu) << 16)); // dim1 hi / tile_dim0=K
    g1[4] = 64;                                                       // tile_dim1=64, tile_dim2=0
    g1[5] = (int)(unsigned)ldw;                                       // tensor_dim0_stride lo32
    g1[6] = 0;                                                        // stride hi / dim1_stride
    g1[7] = 0;
    v4i z4 = {0, 0, 0, 0};
# if __clang_major__ >= 23
    v8i z8 = {0, 0, 0, 0, 0, 0, 0, 0};
    __builtin_amdgcn_tensor_load_to_lds(g0, g1, z4, z4, z8, 0);
# else
    __builtin_amdgcn_tensor_load_to_lds(g0, g1, z4, z4, 0);
# endif
    __builtin_amdgcn_s_wait_tensorcnt((short)0);
  }
#else
  {
    const int kv = K / 8;
    const int nv = 64 * kv;
    for (int i = threadIdx.x; i < nv; i += 128) {
      int r_ = i / kv, c_ = i - r_ * kv;
      ((v8h*)bpanel)[i] = *(const v8h*)(W + (size_t)(col0 + r_) * ldw + c_ * 8);
    }
  }
#endif
  __syncthreads();

  const _Float16* xrow = X + (size_t)(row0 + m) * ldx + kbA;
  const _Float16* bp0  = bpanel + (size_t)( 0 + m) * K + kbB;
  const _Float16* bp1  = bpanel + (size_t)(16 + m) * K + kbB;
  const _Float16* bp2  = bpanel + (size_t)(32 + m) * K + kbB;
  const _Float16* bp3  = bpanel + (size_t)(48 + m) * K + kbB;

  v8f acc[4] = {};
  for (int k = 0; k < K; k += 32) {
    v16h a  = load_afrag(xrow + k);
    v16h b0 = *(const v16h*)(bp0 + k);
    v16h b1 = *(const v16h*)(bp1 + k);
    v16h b2 = *(const v16h*)(bp2 + k);
    v16h b3 = *(const v16h*)(bp3 + k);
    acc[0] = __builtin_amdgcn_wmma_f32_16x16x32_f16(false, a, false, b0, (short)0, acc[0], false, false);
    acc[1] = __builtin_amdgcn_wmma_f32_16x16x32_f16(false, a, false, b1, (short)0, acc[1], false, false);
    acc[2] = __builtin_amdgcn_wmma_f32_16x16x32_f16(false, a, false, b2, (short)0, acc[2], false, false);
    acc[3] = __builtin_amdgcn_wmma_f32_16x16x32_f16(false, a, false, b3, (short)0, acc[3], false, false);
  }

  #pragma unroll
  for (int t = 0; t < 4; ++t) {
    int col = col0 + 16 * t + m;
    float bb = bias ? bias[col] : 0.0f;
    #pragma unroll
    for (int i = 0; i < 8; ++i) {
      int row = row0 + i + 8 * hi;           // D layout: VGPR i -> M = i + 8*(L>=16)
      float v = acc[t][i] + bb;
      if      (act == 1) v = v > 0.0f ? v : 0.0f;
      else if (act == 2) v = 1.0f / (1.0f + __expf(-v));
      else if (act == 3) v = tanhf(v);
      if (Cf) Cf[(size_t)row * ldc  + col] = v;
      if (Ch) Ch[(size_t)row * ldch + col] = (_Float16)v;
    }
  }
}

// ---------------------------------------------------------------------------
// Fused attention: one wave per (16 query rows, head). Full score row-block in
// LDS (16*2048 f32 = 128KB), exact softmax, then P@V with register-built P.
// qh/kh: N_NODE x DIM f16 ; vT: DIM x N_NODE f16 (transposed V) ; atth out.
// ---------------------------------------------------------------------------
__global__ __launch_bounds__(32)
void attn_kernel(const _Float16* __restrict__ qh,
                 const _Float16* __restrict__ kh,
                 const _Float16* __restrict__ vT,
                 _Float16* __restrict__ atth)
{
  extern __shared__ float S[];          // 16*N_NODE scores + 64 reduction slots
  float* red  = S + 16 * N_NODE;        // [32]
  float* mxr  = red + 32;               // [16]
  float* invr = mxr + 16;               // [16]

  const int lane = threadIdx.x;
  const int q0   = blockIdx.x * 16;
  const int h    = blockIdx.y;
  const int m    = lane & 15;
  const int hi   = lane >> 4;
  const float scale = 0.17677669529663687f;   // 1/sqrt(32)

  // Q fragment (reused for every key tile)
  v16h qa = load_afrag(qh + (size_t)(q0 + m) * DIM + h * DK + hi * 16);

  // scores: one WMMA per 16 keys
  for (int j0 = 0; j0 < N_NODE; j0 += 16) {
    v16h b = *(const v16h*)(kh + (size_t)(j0 + m) * DIM + h * DK + hi * 16);
    v8f c = {};
    c = __builtin_amdgcn_wmma_f32_16x16x32_f16(false, qa, false, b, (short)0, c, false, false);
    #pragma unroll
    for (int i = 0; i < 8; ++i)
      S[(size_t)(i + 8 * hi) * N_NODE + j0 + m] = c[i];
  }
  __syncthreads();

  // row max and exp-sum: two lanes cooperate per row
  {
    const int c0 = hi * (N_NODE / 2);
    float lm = -1e30f;
    for (int c = 0; c < N_NODE / 2; ++c)
      lm = fmaxf(lm, S[(size_t)m * N_NODE + c0 + c]);
    red[lane] = lm;
    __syncthreads();
    if (lane < 16) mxr[lane] = fmaxf(red[lane], red[lane + 16]);
    __syncthreads();
    float mx = mxr[m];
    float ls = 0.0f;
    for (int c = 0; c < N_NODE / 2; ++c)
      ls += __expf(scale * (S[(size_t)m * N_NODE + c0 + c] - mx));
    red[lane] = ls;
    __syncthreads();
    if (lane < 16) invr[lane] = 1.0f / (red[lane] + red[lane + 16]);
    __syncthreads();
  }
  const float mx  = mxr[m];
  const float inv = invr[m];

  // P @ V : per 32-key chunk build the P A-fragment in registers
  v8f a0 = {}, a1 = {};
  for (int c0 = 0; c0 < N_NODE; c0 += 32) {
    AFrag pf;
    #pragma unroll
    for (int e = 0; e < 8; ++e) {
      float s0 = S[(size_t)m * N_NODE + c0 + hi * 16 + e];
      float s1 = S[(size_t)m * N_NODE + c0 + hi * 16 + 8 + e];
      pf.h[0][e] = (_Float16)(__expf(scale * (s0 - mx)) * inv);
      pf.h[1][e] = (_Float16)(__expf(scale * (s1 - mx)) * inv);
    }
    v16h b0 = *(const v16h*)(vT + (size_t)(h * DK      + m) * N_NODE + c0 + hi * 16);
    v16h b1 = *(const v16h*)(vT + (size_t)(h * DK + 16 + m) * N_NODE + c0 + hi * 16);
    a0 = __builtin_amdgcn_wmma_f32_16x16x32_f16(false, pf.v, false, b0, (short)0, a0, false, false);
    a1 = __builtin_amdgcn_wmma_f32_16x16x32_f16(false, pf.v, false, b1, (short)0, a1, false, false);
  }
  #pragma unroll
  for (int i = 0; i < 8; ++i) {
    int row = q0 + i + 8 * hi;
    atth[(size_t)row * DIM + h * DK +      m] = (_Float16)a0[i];
    atth[(size_t)row * DIM + h * DK + 16 + m] = (_Float16)a1[i];
  }
}

// ---------------------------------------------------------------------------
// LayerNorm(a + b) with affine; optional f32/f16 outputs. One row per block.
// ---------------------------------------------------------------------------
__global__ __launch_bounds__(256)
void ln_kernel(const float* __restrict__ a, const float* __restrict__ b,
               const float* __restrict__ g, const float* __restrict__ be,
               float* __restrict__ y, _Float16* __restrict__ yh)
{
  __shared__ float sm[256];
  const int row = blockIdx.x, t = threadIdx.x;
  float v = a[(size_t)row * DIM + t] + b[(size_t)row * DIM + t];
  sm[t] = v; __syncthreads();
  for (int s = 128; s > 0; s >>= 1) { if (t < s) sm[t] += sm[t + s]; __syncthreads(); }
  float mean = sm[0] / DIM;
  __syncthreads();
  float d = v - mean;
  sm[t] = d * d; __syncthreads();
  for (int s = 128; s > 0; s >>= 1) { if (t < s) sm[t] += sm[t + s]; __syncthreads(); }
  float var = sm[0] / DIM;
  float o = d * rsqrtf(var + LN_EPS) * g[t] + be[t];
  if (y)  y [(size_t)row * DIM + t] = o;
  if (yh) yh[(size_t)row * DIM + t] = (_Float16)o;
}

// ---------------------------------------------------------------------------
// Elementwise helpers
// ---------------------------------------------------------------------------
__global__ void convert_h(const float* __restrict__ src, _Float16* __restrict__ dst, int n) {
  int i = blockIdx.x * blockDim.x + threadIdx.x;
  if (i < n) dst[i] = (_Float16)src[i];
}

__global__ void convert_adj(const float* __restrict__ A,
                            _Float16* __restrict__ ain, _Float16* __restrict__ aout, int n) {
  int i = blockIdx.x * blockDim.x + threadIdx.x;
  if (i < n) {
    ain [i] = (_Float16)A[2 * (size_t)i];
    aout[i] = (_Float16)A[2 * (size_t)i + 1];
  }
}

__global__ void transpose_h(const _Float16* __restrict__ in, int ldin,
                            _Float16* __restrict__ out, int R, int C) {
  int i = blockIdx.x * blockDim.x + threadIdx.x;
  if (i >= R * C) return;
  int r = i / C, c = i % C;
  out[(size_t)c * R + r] = in[(size_t)r * ldin + c];
}

__global__ void build_cat2(const _Float16* __restrict__ cat1,
                           const float* __restrict__ r, const float* __restrict__ s,
                           _Float16* __restrict__ cat2, int n) {
  int i = blockIdx.x * blockDim.x + threadIdx.x;
  if (i >= n) return;
  int row = i / 768, col = i % 768;
  if (col < 512) cat2[i] = cat1[i];
  else {
    int c = col - 512;
    cat2[i] = (_Float16)(r[(size_t)row * DIM + c] * s[(size_t)row * DIM + c]);
  }
}

__global__ void gru_kernel(const float* __restrict__ z, const float* __restrict__ s,
                           const float* __restrict__ hh,
                           float* __restrict__ xo, _Float16* __restrict__ xh, int n) {
  int i = blockIdx.x * blockDim.x + threadIdx.x;
  if (i < n) {
    float zz = z[i];
    float v = (1.0f - zz) * s[i] + zz * hh[i];
    xo[i] = v;
    xh[i] = (_Float16)v;
  }
}

// ---------------------------------------------------------------------------
extern "C" void kernel_launch(void* const* d_in, const int* in_sizes, int n_in,
                              void* d_out, int out_size, void* d_ws, size_t ws_size,
                              hipStream_t stream)
{
  (void)in_sizes; (void)n_in; (void)out_size; (void)ws_size;
  const int ND = N_NODE * DIM;

  const float* x0  = (const float*)d_in[0];
  const float* A   = (const float*)d_in[1];
  const float* Wq  = (const float*)d_in[2];
  const float* Wk  = (const float*)d_in[3];
  const float* Wv  = (const float*)d_in[4];
  const float* Wo  = (const float*)d_in[5];
  const float* bo  = (const float*)d_in[6];
  const float* Wf1 = (const float*)d_in[7];
  const float* bf1 = (const float*)d_in[8];
  const float* Wf2 = (const float*)d_in[9];
  const float* bf2 = (const float*)d_in[10];
  const float* g1  = (const float*)d_in[11];
  const float* be1 = (const float*)d_in[12];
  const float* g2  = (const float*)d_in[13];
  const float* be2 = (const float*)d_in[14];
  const float* Win = (const float*)d_in[15];
  const float* bin = (const float*)d_in[16];
  // d_in[17]=Wout, d_in[18]=bout : dead in reference (a_out uses state_in)
  const float* Wr  = (const float*)d_in[19];
  const float* br  = (const float*)d_in[20];
  const float* Wz  = (const float*)d_in[21];
  const float* bz  = (const float*)d_in[22];
  const float* Wt  = (const float*)d_in[23];
  const float* bt  = (const float*)d_in[24];

  // --- workspace carve-up (256B aligned) ---
  char* p = (char*)d_ws;
  auto alloc = [&](size_t bytes) -> void* {
    void* r = (void*)p; p += (bytes + 255) & ~(size_t)255; return r;
  };
  _Float16* Ain_h  = (_Float16*)alloc((size_t)N_NODE * N_NODE * 2);
  _Float16* Aout_h = (_Float16*)alloc((size_t)N_NODE * N_NODE * 2);
  _Float16* x_h    = (_Float16*)alloc((size_t)ND * 2);
  _Float16* q_h    = (_Float16*)alloc((size_t)ND * 2);
  _Float16* k_h    = (_Float16*)alloc((size_t)ND * 2);
  _Float16* v_h    = (_Float16*)alloc((size_t)ND * 2);
  _Float16* vT     = (_Float16*)alloc((size_t)ND * 2);
  _Float16* atr_h  = (_Float16*)alloc((size_t)ND * 2);
  float*    att_f  = (float*)   alloc((size_t)ND * 4);
  _Float16* x1_h   = (_Float16*)alloc((size_t)ND * 2);
  _Float16* ff1_h  = (_Float16*)alloc((size_t)N_NODE * 512 * 2);
  float*    ff2_f  = (float*)   alloc((size_t)ND * 4);
  float*    s_f    = (float*)   alloc((size_t)ND * 4);
  _Float16* s_h    = (_Float16*)alloc((size_t)ND * 2);
  _Float16* tA_h   = (_Float16*)alloc((size_t)ND * 2);
  _Float16* tB_h   = (_Float16*)alloc((size_t)ND * 2);
  _Float16* cat1_h = (_Float16*)alloc((size_t)N_NODE * 768 * 2);
  _Float16* cat2_h = (_Float16*)alloc((size_t)N_NODE * 768 * 2);
  _Float16* siT    = (_Float16*)alloc((size_t)ND * 2);
  float*    r_f    = (float*)   alloc((size_t)ND * 4);
  float*    z_f    = (float*)   alloc((size_t)ND * 4);
  float*    hh_f   = (float*)   alloc((size_t)ND * 4);
  float*    xbuf   = (float*)   alloc((size_t)ND * 4);
  _Float16* Wq_h   = (_Float16*)alloc((size_t)DIM * DIM * 2);
  _Float16* Wk_h   = (_Float16*)alloc((size_t)DIM * DIM * 2);
  _Float16* Wv_h   = (_Float16*)alloc((size_t)DIM * DIM * 2);
  _Float16* Wo_h   = (_Float16*)alloc((size_t)DIM * DIM * 2);
  _Float16* Wf1_h  = (_Float16*)alloc((size_t)512 * DIM * 2);
  _Float16* Wf2_h  = (_Float16*)alloc((size_t)DIM * 512 * 2);
  _Float16* Win_h  = (_Float16*)alloc((size_t)4 * DIM * DIM * 2);
  _Float16* Wr_h   = (_Float16*)alloc((size_t)DIM * 768 * 2);
  _Float16* Wz_h   = (_Float16*)alloc((size_t)DIM * 768 * 2);
  _Float16* Wt_h   = (_Float16*)alloc((size_t)DIM * 768 * 2);

  auto CVT = [&](const float* s_, _Float16* d_, int n) {
    convert_h<<<(n + 255) / 256, 256, 0, stream>>>(s_, d_, n);
  };
  auto GEMM = [&](const _Float16* X, int ldx, const _Float16* W, int ldw,
                  const float* bias, float* Cf, int ldc, _Float16* Ch, int ldch,
                  int K, int M, int act) {
    size_t sm = (size_t)64 * K * sizeof(_Float16);   // B panel in LDS (TDM-staged)
    gemm_xwt<<<dim3(M / 64, N_NODE / 64), dim3(128), sm, stream>>>(
        X, ldx, W, ldw, M, bias, Cf, ldc, Ch, ldch, K, act);
  };

  // --- one-time conversions (re-done every call: deterministic) ---
  convert_adj<<<(N_NODE * N_NODE + 255) / 256, 256, 0, stream>>>(A, Ain_h, Aout_h, N_NODE * N_NODE);
  CVT(Wq,  Wq_h,  DIM * DIM);   CVT(Wk,  Wk_h,  DIM * DIM);
  CVT(Wv,  Wv_h,  DIM * DIM);   CVT(Wo,  Wo_h,  DIM * DIM);
  CVT(Wf1, Wf1_h, 512 * DIM);   CVT(Wf2, Wf2_h, DIM * 512);
  CVT(Win, Win_h, 4 * DIM * DIM);
  CVT(Wr,  Wr_h,  DIM * 768);   CVT(Wz,  Wz_h,  DIM * 768);
  CVT(Wt,  Wt_h,  DIM * 768);
  CVT(x0, x_h, ND);

  const size_t smem_attn = (16 * N_NODE + 64) * sizeof(float);   // 128KB scores + reductions

  for (int step = 0; step < 2; ++step) {
    const float* xin = (step == 0) ? x0 : xbuf;

    // --- attention block ---
    GEMM(x_h, DIM, Wq_h, DIM, nullptr, nullptr, 0, q_h, DIM, DIM, DIM, 0);
    GEMM(x_h, DIM, Wk_h, DIM, nullptr, nullptr, 0, k_h, DIM, DIM, DIM, 0);
    GEMM(x_h, DIM, Wv_h, DIM, nullptr, nullptr, 0, v_h, DIM, DIM, DIM, 0);
    transpose_h<<<(ND + 255) / 256, 256, 0, stream>>>(v_h, DIM, vT, N_NODE, DIM);
    attn_kernel<<<dim3(N_NODE / 16, NH), dim3(32), smem_attn, stream>>>(q_h, k_h, vT, atr_h);
    GEMM(atr_h, DIM, Wo_h, DIM, bo, att_f, DIM, nullptr, 0, DIM, DIM, 0);
    ln_kernel<<<N_NODE, 256, 0, stream>>>(xin, att_f, g1, be1, nullptr, x1_h);
    GEMM(x1_h, DIM, Wf1_h, DIM, bf1, nullptr, 0, ff1_h, 512, DIM, 512, 1);       // relu
    GEMM(ff1_h, 512, Wf2_h, 512, bf2, ff2_f, DIM, nullptr, 0, 512, DIM, 0);
    ln_kernel<<<N_NODE, 256, 0, stream>>>(ff2_f, att_f, g2, be2, s_f, s_h);      // s

    // --- Win chain: state_in (state_out is dead in the reference) ---
    const _Float16* cin = s_h;
    for (int i = 0; i < 4; ++i) {
      _Float16* coutH = (i == 3) ? (cat1_h + 512) : ((i & 1) ? tB_h : tA_h);
      int ldo = (i == 3) ? 768 : DIM;
      GEMM(cin, DIM, Win_h + (size_t)i * DIM * DIM, DIM, bin + i * DIM,
           nullptr, 0, coutH, ldo, DIM, DIM, 0);
      cin = coutH;
    }

    // --- adjacency matmuls: a_in / a_out into concat buffer ---
    transpose_h<<<(ND + 255) / 256, 256, 0, stream>>>(cat1_h + 512, 768, siT, N_NODE, DIM);
    GEMM(Ain_h,  N_NODE, siT, N_NODE, nullptr, nullptr, 0, cat1_h,       768, N_NODE, DIM, 0);
    GEMM(Aout_h, N_NODE, siT, N_NODE, nullptr, nullptr, 0, cat1_h + 256, 768, N_NODE, DIM, 0);

    // --- gates ---
    GEMM(cat1_h, 768, Wr_h, 768, br, r_f, DIM, nullptr, 0, 768, DIM, 2);         // sigmoid
    GEMM(cat1_h, 768, Wz_h, 768, bz, z_f, DIM, nullptr, 0, 768, DIM, 2);         // sigmoid
    build_cat2<<<(N_NODE * 768 + 255) / 256, 256, 0, stream>>>(cat1_h, r_f, s_f, cat2_h, N_NODE * 768);
    GEMM(cat2_h, 768, Wt_h, 768, bt, hh_f, DIM, nullptr, 0, 768, DIM, 3);        // tanh

    float* xout = (step == 1) ? (float*)d_out : xbuf;
    gru_kernel<<<(ND + 255) / 256, 256, 0, stream>>>(z_f, s_f, hh_f, xout, x_h, ND);
  }
}
